// TerminalODEBlock_75977971466307
// MI455X (gfx1250) — matrix operand order repaired and checked by reading, not compiled, not run on hardware
//
#include <hip/hip_runtime.h>
#include <math.h>

// ---------------------------------------------------------------------------
// TerminalODEBlock: 32-step RK4 over f(s,h) = W3·tanh(W2·tanh(W1·[s,h]+b1)+b2)+b3
// B=4096, D=1024, IN=1025 (padded to 1088), H1=H2=4096.
// bf16 WMMA GEMMs (f32 accum); weights converted+transposed once (L2-resident);
// double-buffered LDS fed by GLOBAL_LOAD_ASYNC_TO_LDS_B128 (ASYNCcnt pipeline).
// ---------------------------------------------------------------------------

typedef __bf16 bf16_t;
typedef __bf16 v16bf __attribute__((ext_vector_type(16)));
typedef __bf16 v8bf  __attribute__((ext_vector_type(8)));
typedef float  v8f   __attribute__((ext_vector_type(8)));
typedef int    v4i_t __attribute__((vector_size(16)));   // builtin pointee type

#define B_SZ   4096
#define D_SZ   1024
#define IN_SZ  1025
#define KP1    1088      // IN padded to multiple of 64
#define H1_SZ  4096
#define H2_SZ  4096
#define NSTEPS 32        // matches reference setup_inputs()

#define KCH    64        // K-chunk per LDS stage
#define LDS_LD 72        // padded LDS row stride in halves (144B: bank spread)

#if defined(__AMDGCN__) && __has_builtin(__builtin_amdgcn_global_load_async_to_lds_b128)
#define HAVE_ASYNC_LDS 1
#else
#define HAVE_ASYNC_LDS 0
#endif

#if defined(__AMDGCN__) && __has_builtin(__builtin_amdgcn_s_wait_asynccnt)
#define ASYNC_WAIT(n) __builtin_amdgcn_s_wait_asynccnt(n)
#elif HAVE_ASYNC_LDS
#define ASYNC_WAIT(n) asm volatile("s_wait_asynccnt %0" ::"n"(n) : "memory")
#else
#define ASYNC_WAIT(n) ((void)0)
#endif

__device__ __forceinline__ void async_ld16(const void* g, void* l)
{
#if HAVE_ASYNC_LDS
    __builtin_amdgcn_global_load_async_to_lds_b128(
        (__attribute__((address_space(1))) v4i_t*)(g),
        (__attribute__((address_space(3))) v4i_t*)(l), 0, 0);
#else
    *(uint4*)l = *(const uint4*)g;   // sync fallback (correct, slower)
#endif
}

// ---------------------------------------------------------------------------
// Tiled bf16 WMMA GEMM:  out = act(A[M,K] * Bt[N,K]^T + bias)
//   A  : bf16 row-major, lda = K;  Bt : row n = column n of W (contig in K)
//   f32out==0 -> outb = bf16(tanh(.)) ; f32out==1 -> outf = f32(.) (no tanh)
// Block: 256 threads = 8 wave32s as 2(M)x4(N); tile 128x128; K-chunk 64;
// double-buffered async-to-LDS staging.
// ---------------------------------------------------------------------------
__global__ __launch_bounds__(256)
void k_gemm_bf16(const bf16_t* __restrict__ A, const bf16_t* __restrict__ Bt,
                 const float* __restrict__ bias, bf16_t* __restrict__ outb,
                 float* __restrict__ outf, int N, int K, int f32out)
{
    __shared__ __align__(16) bf16_t sA[2 * 128 * LDS_LD];
    __shared__ __align__(16) bf16_t sB[2 * 128 * LDS_LD];

    const int tid  = threadIdx.x;
    const int lane = tid & 31;
    const int wave = tid >> 5;
    const int wm   = wave >> 2;          // 0..1 : 64 rows per wave
    const int wn   = wave & 3;           // 0..3 : 32 cols per wave
    const int mBase = blockIdx.y * 128;
    const int nBase = blockIdx.x * 128;

    // staging: thread -> tile row r, half-row segment seg (32 halves = 64B)
    const int r   = tid >> 1;
    const int seg = tid & 1;
    const bf16_t* gA = A  + (long)(mBase + r) * K + seg * 32;
    const bf16_t* gB = Bt + (long)(nBase + r) * K + seg * 32;
    bf16_t* lA = &sA[r * LDS_LD + seg * 32];
    bf16_t* lB = &sB[r * LDS_LD + seg * 32];

    auto stage = [&](int buf, int kc) {
        const bf16_t* ga = gA + kc;
        const bf16_t* gb = gB + kc;
        bf16_t* la = lA + buf * (128 * LDS_LD);
        bf16_t* lb = lB + buf * (128 * LDS_LD);
        async_ld16(ga,      la);
        async_ld16(ga + 8,  la + 8);
        async_ld16(ga + 16, la + 16);
        async_ld16(ga + 24, la + 24);
        async_ld16(gb,      lb);
        async_ld16(gb + 8,  lb + 8);
        async_ld16(gb + 16, lb + 16);
        async_ld16(gb + 24, lb + 24);
    };

    v8f acc[4][2] = {};

    const int hi = lane >> 4;            // lane group (K-interleave select)
    const int ml = lane & 15;            // row (A) / col (B/D) within 16

    const int nch = K / KCH;
    stage(0, 0);                          // prologue: chunk 0 -> buffer 0
    for (int i = 0; i < nch; ++i) {
        const int cur = i & 1;
        if (i + 1 < nch) {                // keep next chunk in flight
            stage(cur ^ 1, (i + 1) * KCH);
            ASYNC_WAIT(8);                // chunk i landed; chunk i+1 in flight
        } else {
            ASYNC_WAIT(0);
        }
        __syncthreads();

        const bf16_t* bufA = &sA[cur * (128 * LDS_LD)];
        const bf16_t* bufB = &sB[cur * (128 * LDS_LD)];
#pragma unroll
        for (int kk = 0; kk < KCH; kk += 32) {
            // A frags: 16x32, lane ml = row; elem j -> k = 16*(j>>3)+8*hi+(j&7)
            v16bf afr[4];
#pragma unroll
            for (int mi = 0; mi < 4; ++mi) {
                const bf16_t* p = bufA + (wm * 64 + mi * 16 + ml) * LDS_LD + kk + 8 * hi;
                v8bf lo = *(const v8bf*)p;
                v8bf h8 = *(const v8bf*)(p + 16);
                afr[mi] = __builtin_shufflevector(lo, h8,
                            0,1,2,3,4,5,6,7,8,9,10,11,12,13,14,15);
            }
            // B frags: 32x16, lane ml = col; elem j -> k = 16*(j>>3)+8*hi+(j&7)
            v16bf bfr[2];
#pragma unroll
            for (int ni = 0; ni < 2; ++ni) {
                const bf16_t* p = bufB + (wn * 32 + ni * 16 + ml) * LDS_LD + kk + 8 * hi;
                v8bf lo = *(const v8bf*)p;
                v8bf h8 = *(const v8bf*)(p + 16);
                bfr[ni] = __builtin_shufflevector(lo, h8,
                            0,1,2,3,4,5,6,7,8,9,10,11,12,13,14,15);
            }
#pragma unroll
            for (int mi = 0; mi < 4; ++mi)
#pragma unroll
                for (int ni = 0; ni < 2; ++ni)
                    acc[mi][ni] = __builtin_amdgcn_wmma_f32_16x16x32_bf16(
                        false, afr[mi], false, bfr[ni],
                        (short)0, acc[mi][ni], false, false);
        }
        __syncthreads();
    }

    // Epilogue. D layout: lane ml = col; VGPR rr -> row rr + 8*hi.
#pragma unroll
    for (int mi = 0; mi < 4; ++mi) {
#pragma unroll
        for (int ni = 0; ni < 2; ++ni) {
            int col  = nBase + wn * 32 + ni * 16 + ml;
            float bv = bias[col];
#pragma unroll
            for (int rr = 0; rr < 8; ++rr) {
                int row = mBase + wm * 64 + mi * 16 + hi * 8 + rr;
                float v = acc[mi][ni][rr] + bv;
                if (f32out) outf[(long)row * N + col] = v;
                else        outb[(long)row * N + col] = (bf16_t)tanhf(v);
            }
        }
    }
}

// --------------------------- elementwise helpers ---------------------------

// W[K,N] fp32 row-major -> Wt[N,Kpad] bf16 (zero-padded K tail)
__global__ __launch_bounds__(256)
void k_convert_wt(const float* __restrict__ W, bf16_t* __restrict__ Wt,
                  int K, int N, int Kpad, long total)
{
    long idx = (long)blockIdx.x * 256 + threadIdx.x;
    if (idx >= total) return;
    int n = (int)(idx / Kpad);
    int k = (int)(idx % Kpad);
    float v = (k < K) ? W[(long)k * N + n] : 0.0f;
    Wt[idx] = (bf16_t)v;
}

// x = bf16([s, h + c*dt*kprev, 0-pad]); dt computed on-device (capture-safe)
__global__ __launch_bounds__(256)
void k_build_x(const float* __restrict__ h, const float* __restrict__ kprev,
               const float* __restrict__ tN, const float* __restrict__ tNp1,
               float stepf, float sfrac, float cfrac, bf16_t* __restrict__ xbf)
{
    long idx = (long)blockIdx.x * 256 + threadIdx.x;
    long total = (long)B_SZ * KP1;
    if (idx >= total) return;
    int b = (int)(idx / KP1);
    int j = (int)(idx % KP1);
    float dt = (tNp1[0] - tN[0]) * (1.0f / (float)NSTEPS);
    float v;
    if (j == 0)         v = (stepf + sfrac) * dt;
    else if (j <= D_SZ) v = h[(long)b * D_SZ + (j - 1)]
                            + cfrac * dt * kprev[(long)b * D_SZ + (j - 1)];
    else                v = 0.0f;
    xbf[idx] = (bf16_t)v;
}

__global__ __launch_bounds__(256)
void k_copy(float* __restrict__ dst, const float* __restrict__ src, long n)
{
    long idx = (long)blockIdx.x * 256 + threadIdx.x;
    if (idx < n) dst[idx] = src[idx];
}

__global__ __launch_bounds__(256)
void k_axpy(float* __restrict__ acc, const float* __restrict__ k,
            float w, int init, long n)
{
    long idx = (long)blockIdx.x * 256 + threadIdx.x;
    if (idx >= n) return;
    float kv = w * k[idx];
    acc[idx] = init ? kv : (acc[idx] + kv);
}

__global__ __launch_bounds__(256)
void k_update_h(float* __restrict__ h, const float* __restrict__ acc,
                const float* __restrict__ tN, const float* __restrict__ tNp1,
                long n)
{
    long idx = (long)blockIdx.x * 256 + threadIdx.x;
    if (idx >= n) return;
    float dt = (tNp1[0] - tN[0]) * (1.0f / (float)NSTEPS);
    h[idx] += dt * (1.0f / 6.0f) * acc[idx];
}

__global__ __launch_bounds__(256)
void k_final(float* __restrict__ out, const float* __restrict__ h,
             const float* __restrict__ hN, const float* __restrict__ tN,
             const float* __restrict__ tNp1, long n)
{
    long idx = (long)blockIdx.x * 256 + threadIdx.x;
    if (idx >= n) return;
    int b = (int)(idx / D_SZ);
    float d = tNp1[b] - tN[b];
    out[idx] = (d > 0.0f) ? h[idx] : hN[idx];
}

// ------------------------------ host driver --------------------------------

extern "C" void kernel_launch(void* const* d_in, const int* in_sizes, int n_in,
                              void* d_out, int out_size, void* d_ws, size_t ws_size,
                              hipStream_t stream)
{
    const float* hN  = (const float*)d_in[0];
    const float* tN  = (const float*)d_in[1];
    const float* tNp = (const float*)d_in[2];
    const float* W1  = (const float*)d_in[3];
    const float* b1  = (const float*)d_in[4];
    const float* W2  = (const float*)d_in[5];
    const float* b2  = (const float*)d_in[6];
    const float* W3  = (const float*)d_in[7];
    const float* b3  = (const float*)d_in[8];
    (void)in_sizes; (void)n_in; (void)out_size; (void)ws_size;

    char* ws = (char*)d_ws;
    size_t off = 0;
    auto carve = [&](size_t bytes) -> char* {
        char* p = ws + off;
        off = (off + bytes + 255) & ~(size_t)255;
        return p;
    };
    bf16_t* W1t  = (bf16_t*)carve((size_t)H1_SZ * KP1   * 2);
    bf16_t* W2t  = (bf16_t*)carve((size_t)H2_SZ * H1_SZ * 2);
    bf16_t* W3t  = (bf16_t*)carve((size_t)D_SZ  * H2_SZ * 2);
    bf16_t* xbf  = (bf16_t*)carve((size_t)B_SZ  * KP1   * 2);
    bf16_t* a1   = (bf16_t*)carve((size_t)B_SZ  * H1_SZ * 2);
    bf16_t* a2   = (bf16_t*)carve((size_t)B_SZ  * H2_SZ * 2);
    float*  hbuf = (float*) carve((size_t)B_SZ  * D_SZ  * 4);
    float*  accb = (float*) carve((size_t)B_SZ  * D_SZ  * 4);
    float*  kcur = (float*) carve((size_t)B_SZ  * D_SZ  * 4);

    auto blocks1d = [](long total) {
        return dim3((unsigned)((total + 255) / 256));
    };

    // One-time (per launch) weight convert + transpose to bf16 [N, Kpad]
    k_convert_wt<<<blocks1d((long)H1_SZ * KP1),   256, 0, stream>>>(W1, W1t, IN_SZ, H1_SZ, KP1,   (long)H1_SZ * KP1);
    k_convert_wt<<<blocks1d((long)H2_SZ * H1_SZ), 256, 0, stream>>>(W2, W2t, H1_SZ, H2_SZ, H1_SZ, (long)H2_SZ * H1_SZ);
    k_convert_wt<<<blocks1d((long)D_SZ  * H2_SZ), 256, 0, stream>>>(W3, W3t, H2_SZ, D_SZ,  H2_SZ, (long)D_SZ  * H2_SZ);

    const long nh = (long)B_SZ * D_SZ;
    k_copy<<<blocks1d(nh), 256, 0, stream>>>(hbuf, hN, nh);

    const float cfr[4] = {0.0f, 0.5f, 0.5f, 1.0f};  // stage-input k coefficient
    const float sfr[4] = {0.0f, 0.5f, 0.5f, 1.0f};  // time fraction
    const float wgt[4] = {1.0f, 2.0f, 2.0f, 1.0f};  // RK4 weights

    dim3 g1(H1_SZ / 128, B_SZ / 128);   // 32 x 32 blocks
    dim3 g2(H2_SZ / 128, B_SZ / 128);   // 32 x 32
    dim3 g3(D_SZ  / 128, B_SZ / 128);   //  8 x 32

    for (int step = 0; step < NSTEPS; ++step) {
        for (int st = 0; st < 4; ++st) {
            const float* kp = (st == 0) ? hbuf : kcur;   // coef 0 at st==0
            k_build_x<<<blocks1d((long)B_SZ * KP1), 256, 0, stream>>>(
                hbuf, kp, tN, tNp, (float)step, sfr[st], cfr[st], xbf);
            k_gemm_bf16<<<g1, 256, 0, stream>>>(xbf, W1t, b1, a1, nullptr, H1_SZ, KP1,   0);
            k_gemm_bf16<<<g2, 256, 0, stream>>>(a1,  W2t, b2, a2, nullptr, H2_SZ, H1_SZ, 0);
            k_gemm_bf16<<<g3, 256, 0, stream>>>(a2,  W3t, b3, nullptr, kcur, D_SZ,  H2_SZ, 1);
            k_axpy<<<blocks1d(nh), 256, 0, stream>>>(accb, kcur, wgt[st], st == 0, nh);
        }
        k_update_h<<<blocks1d(nh), 256, 0, stream>>>(hbuf, accb, tN, tNp, nh);
    }
    k_final<<<blocks1d(nh), 256, 0, stream>>>((float*)d_out, hbuf, hN, tN, tNp, nh);
}
